// DepthContrastiveCorrelationLoss_31696858645017
// MI455X (gfx1250) — matrix operand order repaired and checked by reading, not compiled, not run on hardware
//
#include <hip/hip_runtime.h>
#include <math.h>

// ---- problem constants ----
#define NB    32
#define CF    768
#define CCODE 70
#define CPAD  96      // code channels padded to multiple of 32 (WMMA K-step)
#define HW    28
#define FS    11
#define NPTS  121
#define PPAD  128     // points padded to multiple of 16 (WMMA M/N tiles)
#define NNEG  5
#define NPAIR 7       // pos_intra, pos_inter, 5 negs

// output layout (floats), in reference return order
#define CD_SZ        468512ULL            // 32*121*121
#define O_PI_SCALAR  0ULL
#define O_PI_CD      1ULL
#define O_PP_SCALAR  468513ULL
#define O_PP_CD      468514ULL
#define O_NEG_LOSS   937026ULL
#define O_NEG_CD     3279586ULL

typedef __attribute__((ext_vector_type(16))) __bf16 v16bf;
typedef __attribute__((ext_vector_type(8)))  __bf16 v8bf;
typedef __attribute__((ext_vector_type(8)))  float  v8f;

__device__ __forceinline__ unsigned short f2bf(float f) {
  union { float f; unsigned u; } v; v.f = f;
  unsigned u = v.u;
  unsigned r = u + 0x7FFFu + ((u >> 16) & 1u);   // round-to-nearest-even
  return (unsigned short)(r >> 16);
}

// ---------------- k0: zero rowsum/gsum/acc (contiguous region) ----------------
__global__ void k0_zero(float* __restrict__ z) {
  int i = blockIdx.x * blockDim.x + threadIdx.x;
  if (i < NPAIR * NB * PPAD + NPAIR + 2) z[i] = 0.f;
}

// ---------------- k1: grid-sample + L2-normalize -> bf16 scratch ----------------
// feature kinds 0..7: 0=feats@c1, 1=daf@c1, 2=feats_pos@c2, 3..7=feats[perm_i]@c2
// code kinds 8..14:   8=code@c1, 9=code_pos@c2, 10..14=code[perm_i]@c2
__global__ void k1_sample(const float* __restrict__ feats,
                          const float* __restrict__ feats_pos,
                          const float* __restrict__ code,
                          const float* __restrict__ code_pos,
                          const float* __restrict__ daf,
                          const float* __restrict__ coords1,
                          const float* __restrict__ coords2,
                          const int*   __restrict__ perms,
                          unsigned short* __restrict__ wsF,
                          unsigned short* __restrict__ wsC) {
  int task = blockIdx.x;
  int p = task & 127; task >>= 7;
  int n = task & 31;  task >>= 5;
  int kind = task;                      // 0..14
  const int tid = threadIdx.x;

  int Creal, Cpad, nb;
  const float* src; const float* co;
  unsigned short* dst;
  if (kind < 8) {
    Creal = CF; Cpad = CF;
    switch (kind) {
      case 0: src = feats;     nb = n; co = coords1; break;
      case 1: src = daf;       nb = n; co = coords1; break;
      case 2: src = feats_pos; nb = n; co = coords2; break;
      default: { int i = kind - 3; src = feats; nb = perms[i * NB + n]; co = coords2; } break;
    }
    dst = wsF + (((size_t)kind * NB + n) * PPAD + p) * CF;
  } else {
    Creal = CCODE; Cpad = CPAD;
    int ck = kind - 8;
    switch (ck) {
      case 0: src = code;     nb = n; co = coords1; break;
      case 1: src = code_pos; nb = n; co = coords2; break;
      default: { int i = ck - 2; src = code; nb = perms[i * NB + n]; co = coords2; } break;
    }
    dst = wsC + (((size_t)(kind - 8) * NB + n) * PPAD + p) * CPAD;
  }

  if (p >= NPTS) {                      // zero pad rows
    for (int c = tid; c < Cpad; c += blockDim.x) dst[c] = 0;
    return;
  }

  // grid = coords.permute(0,2,1,3): output point p=(h,w) reads coords[n, w, h]
  int h = p / FS, w = p % FS;
  const float* cp = co + (((size_t)n * FS + w) * FS + h) * 2;
  float gx = cp[0] * 2.f - 1.f, gy = cp[1] * 2.f - 1.f;
  float x = fminf(fmaxf((gx + 1.f) * 0.5f * (HW - 1), 0.f), (float)(HW - 1));
  float y = fminf(fmaxf((gy + 1.f) * 0.5f * (HW - 1), 0.f), (float)(HW - 1));
  float x0f = floorf(x), y0f = floorf(y);
  int x0 = (int)x0f, y0 = (int)y0f;
  int x1 = min(x0 + 1, HW - 1), y1 = min(y0 + 1, HW - 1);
  float wx = x - x0f, wy = y - y0f;
  float w00 = (1.f - wx) * (1.f - wy), w01 = wx * (1.f - wy);
  float w10 = (1.f - wx) * wy,         w11 = wx * wy;

  float vals[6];                        // up to 768/128 channels per thread
  float ss = 0.f; int nv = 0;
  const float* base = src + (size_t)nb * Creal * (HW * HW);
  for (int c = tid; c < Cpad; c += blockDim.x) {
    float v = 0.f;
    if (c < Creal) {
      const float* img = base + (size_t)c * (HW * HW);
      v = w00 * img[y0 * HW + x0] + w01 * img[y0 * HW + x1]
        + w10 * img[y1 * HW + x0] + w11 * img[y1 * HW + x1];
    }
    vals[nv++] = v; ss += v * v;
  }
  __shared__ float red[128];
  red[tid] = ss; __syncthreads();
  for (int s = 64; s > 0; s >>= 1) { if (tid < s) red[tid] += red[tid + s]; __syncthreads(); }
  __shared__ float rnormS;
  if (tid == 0) rnormS = 1.f / fmaxf(sqrtf(red[0]), 1e-10f);
  __syncthreads();
  float rn = rnormS; nv = 0;
  for (int c = tid; c < Cpad; c += blockDim.x) dst[c] = f2bf(vals[nv++] * rn);
}

// ---------------- k2: WMMA correlations (fd: K=768, cd: K=96) ----------------
// one wave32 per (pair, n, tileM, tileN); fd -> scratch (+row/global sums), cd -> d_out
__global__ void __launch_bounds__(32) k2_corr(const unsigned short* __restrict__ wsF,
                                              const unsigned short* __restrict__ wsC,
                                              float* __restrict__ fdS,
                                              float* __restrict__ rowsum,
                                              float* __restrict__ gsum,
                                              float* __restrict__ out) {
  int idx = blockIdx.x;
  int tn = idx & 7; idx >>= 3;
  int tm = idx & 7; idx >>= 3;
  int n  = idx & 31; idx >>= 5;
  int pair = idx;                              // 0..6
  int lane  = threadIdx.x;
  int laneM = lane & 15;
  int hi    = lane >> 4;                       // 0/1 half-wave select

  int f1k = (pair == 0) ? 1 : 0;               // pos_intra uses daf, rest use feats
  int f2k = pair + 1;                          // 1=daf,2=feats_pos,3..7=feats_neg
  int c2k = pair;                              // 0=code,1=code_pos,2..6=code_neg

  const unsigned short* A  = wsF + ((size_t)f1k * NB + n) * PPAD * CF;
  const unsigned short* Bm = wsF + ((size_t)f2k * NB + n) * PPAD * CF;
  const unsigned short* Ca = wsC + ((size_t)0   * NB + n) * PPAD * CPAD;
  const unsigned short* Cb = wsC + ((size_t)c2k * NB + n) * PPAD * CPAD;

  int rowA = tm * 16 + laneM;                  // output row  (from f1)
  int rowB = tn * 16 + laneM;                  // output col  (from f2, row-major => B^T)

  // fd = f1 . f2^T over K=768
  v8f accF = {0.f,0.f,0.f,0.f,0.f,0.f,0.f,0.f};
  const unsigned short* aPtr = A  + (size_t)rowA * CF + hi * 16;
  const unsigned short* bPtr = Bm + (size_t)rowB * CF + hi * 16;
  for (int kk = 0; kk < CF; kk += 32) {
    union { v16bf v; v8bf h[2]; } ua;
    ua.h[0] = *(const v8bf*)(aPtr + kk);       // K: kk + hi*16 + 0..7
    ua.h[1] = *(const v8bf*)(aPtr + kk + 8);   // K: kk + hi*16 + 8..15
    v16bf b = *(const v16bf*)(bPtr + kk);      // K: kk + hi*16 + 0..15
    accF = __builtin_amdgcn_wmma_f32_16x16x32_bf16(false, ua.v, false, b,
                                                   (short)0, accF, false, false);
  }
  // cd = c1 . c2^T over padded K=96 (channels 70..95 are zero)
  v8f accC = {0.f,0.f,0.f,0.f,0.f,0.f,0.f,0.f};
  const unsigned short* caPtr = Ca + (size_t)rowA * CPAD + hi * 16;
  const unsigned short* cbPtr = Cb + (size_t)rowB * CPAD + hi * 16;
  for (int kk = 0; kk < CPAD; kk += 32) {
    union { v16bf v; v8bf h[2]; } ua;
    ua.h[0] = *(const v8bf*)(caPtr + kk);
    ua.h[1] = *(const v8bf*)(caPtr + kk + 8);
    v16bf b = *(const v16bf*)(cbPtr + kk);
    accC = __builtin_amdgcn_wmma_f32_16x16x32_bf16(false, ua.v, false, b,
                                                   (short)0, accC, false, false);
  }

  float* fdBase = fdS    + ((size_t)pair * NB + n) * PPAD * PPAD;
  float* rsBase = rowsum + ((size_t)pair * NB + n) * PPAD;
  size_t cdBase;
  if      (pair == 0) cdBase = O_PI_CD;
  else if (pair == 1) cdBase = O_PP_CD;
  else                cdBase = O_NEG_CD + (size_t)(pair - 2) * CD_SZ;

  int col = tn * 16 + laneM;
  float lsum = 0.f;
#pragma unroll
  for (int r = 0; r < 8; ++r) {
    int row = tm * 16 + hi * 8 + r;            // C/D layout: lanes>=16 hold M+8
    float fv = accF[r];
    fdBase[(size_t)row * PPAD + col] = fv;     // pad entries are exact zeros
    atomicAdd(rsBase + row, fv);
    lsum += fv;
    if (row < NPTS && col < NPTS)
      out[cdBase + (size_t)n * (NPTS * NPTS) + (size_t)row * NPTS + col] = accC[r];
  }
  atomicAdd(gsum + pair, lsum);
}

// ---------------- k3: finalize loss elementwise ----------------
__global__ void k3_final(const float* __restrict__ fdS,
                         const float* __restrict__ rowsum,
                         const float* __restrict__ gsum,
                         float* __restrict__ acc,
                         float* __restrict__ out) {
  const long long total = (long long)NPAIR * NB * NPTS * NPTS;
  for (long long e = blockIdx.x * (long long)blockDim.x + threadIdx.x; e < total;
       e += (long long)gridDim.x * blockDim.x) {
    int pair = (int)(e / ((long long)NB * NPTS * NPTS));
    long long rem = e % ((long long)NB * NPTS * NPTS);
    int n  = (int)(rem / (NPTS * NPTS));
    int pq = (int)(rem % (NPTS * NPTS));
    int p = pq / NPTS, q = pq % NPTS;

    float fv    = fdS[(((size_t)pair * NB + n) * PPAD + p) * PPAD + q];
    float rmean = rowsum[((size_t)pair * NB + n) * PPAD + p] * (1.f / NPTS);
    float gmean = gsum[pair] * (1.f / (float)CD_SZ);
    float fdf   = fv - rmean + gmean;          // fd - mean_ij(fd) + global_mean(fd)

    size_t cdBase; float shift;
    if      (pair == 0) { cdBase = O_PI_CD; shift = 0.18f; }
    else if (pair == 1) { cdBase = O_PP_CD; shift = 0.12f; }
    else { cdBase = O_NEG_CD + (size_t)(pair - 2) * CD_SZ; shift = 0.46f; }

    float cd = out[cdBase + (size_t)n * (NPTS * NPTS) + pq];
    float loss = -fminf(fmaxf(cd, 0.f), 0.8f) * (fdf - shift);

    if (pair < 2) atomicAdd(acc + pair, loss);
    else out[O_NEG_LOSS + (size_t)(pair - 2) * CD_SZ + (size_t)n * (NPTS * NPTS) + pq] = loss;
  }
}

// ---------------- k4: scalar means ----------------
__global__ void k4_scalars(const float* __restrict__ acc, float* __restrict__ out) {
  out[O_PI_SCALAR] = acc[0] * (1.f / (float)CD_SZ);
  out[O_PP_SCALAR] = acc[1] * (1.f / (float)CD_SZ);
}

extern "C" void kernel_launch(void* const* d_in, const int* in_sizes, int n_in,
                              void* d_out, int out_size, void* d_ws, size_t ws_size,
                              hipStream_t stream) {
  const float* feats     = (const float*)d_in[0];
  const float* feats_pos = (const float*)d_in[1];
  // d_in[2], d_in[3]: salience (unused)
  const float* code      = (const float*)d_in[4];
  const float* code_pos  = (const float*)d_in[5];
  const float* daf       = (const float*)d_in[6];
  // d_in[7]: depth_aug_feats_pos sampled but unused in reference outputs
  const float* coords1   = (const float*)d_in[8];
  const float* coords2   = (const float*)d_in[9];
  const int*   perms     = (const int*)d_in[10];
  float* out = (float*)d_out;

  // workspace layout: bf16 feats | bf16 code | f32 fd | f32 rowsum | f32 gsum | f32 acc
  unsigned short* wsF = (unsigned short*)d_ws;                       // 8*32*128*768
  unsigned short* wsC = wsF + (size_t)8 * NB * PPAD * CF;            // 7*32*128*96
  float* fdS    = (float*)(wsC + (size_t)NPAIR * NB * PPAD * CPAD);  // 7*32*128*128
  float* rowsum = fdS + (size_t)NPAIR * NB * PPAD * PPAD;            // 7*32*128
  float* gsum   = rowsum + NPAIR * NB * PPAD;                        // 7
  float* acc    = gsum + NPAIR;                                      // 2

  int nz = NPAIR * NB * PPAD + NPAIR + 2;
  hipLaunchKernelGGL(k0_zero, dim3((nz + 255) / 256), dim3(256), 0, stream, rowsum);
  hipLaunchKernelGGL(k1_sample, dim3(15 * NB * PPAD), dim3(128), 0, stream,
                     feats, feats_pos, code, code_pos, daf, coords1, coords2, perms, wsF, wsC);
  hipLaunchKernelGGL(k2_corr, dim3(NPAIR * NB * 64), dim3(32), 0, stream,
                     wsF, wsC, fdS, rowsum, gsum, out);
  long long total3 = (long long)NPAIR * NB * NPTS * NPTS;
  hipLaunchKernelGGL(k3_final, dim3((unsigned)((total3 + 255) / 256)), dim3(256), 0, stream,
                     fdS, rowsum, gsum, acc, out);
  hipLaunchKernelGGL(k4_scalars, dim3(1), dim3(1), 0, stream, acc, out);
}
